// QuantumConvLayer_55671366091163
// MI455X (gfx1250) — matrix operand order repaired and checked by reading, not compiled, not run on hardware
//
#include <hip/hip_runtime.h>

typedef unsigned short u16;
typedef unsigned int u32;

typedef __attribute__((ext_vector_type(16))) __bf16 v16bf;
typedef __attribute__((ext_vector_type(8)))  float  v8f;
typedef __attribute__((ext_vector_type(8)))  int    v8i;

#define BATCH 16
#define CIN   64
#define HWIN  256
#define HO    254
#define WO    254
#define COUT  64
#define KTOT  768            // 64 c * 3 kh * 4 (kw padded 3->4)
#define NGI   192            // rows: c*3+kh
#define CHUNKS 24            // KTOT / 32
#define WT    128            // spatial tile (w) per block
#define PITCH 136            // LDS row pitch in bf16 elems (stage 132 used)
#define XS1   (NGI*PITCH + 16)          // ushort offset of shifted copy (+32B skew)
#define PS_OFF ((XS1 + NGI*PITCH) * 2)  // byte offset of partial-sum scratch in LDS
#define SMEM_BYTES (PS_OFF + 4096)
#define NBLK  (2*HO*BATCH)   // conv blocks = 8128
#define NPIX  (BATCH*HO*WO)  // per-channel element count = 1032256
#define PLANE (HO*WO)        // 64516

__device__ __forceinline__ u16 f2bf(float f) {
  u32 u = __float_as_uint(f);
  return (u16)((u + 0x7fffu + ((u >> 16) & 1u)) >> 16);
}

// ---------------------------------------------------------------------------
// Kernel 1: fold quantum circuit + 1x1 conv into effective 3x3 weights (bf16,
// k-major with kw padded to 4 slots; slot 3 weight = 0 so the extra element a
// B-fragment dword drags in is multiplied by zero).
// ---------------------------------------------------------------------------
__global__ void prep_kernel(const float* __restrict__ w_pre,
                            const float* __restrict__ qp,
                            const float* __restrict__ w_post,
                            u16* __restrict__ Wb) {
  float s0[4], s1[4];
  for (int q = 0; q < 4; ++q) {
    s0[q] = __cosf((qp[q*3+0] + qp[q*3+1]) * 0.5f) * __cosf(qp[q*3+2]);
    s1[q] = __cosf((qp[12+q*3+0] + qp[12+q*3+1]) * 0.5f) * __cosf(qp[12+q*3+2]);
  }
  float A2[4][4], T[4][4];
  for (int q = 0; q < 4; ++q)
    for (int p = 0; p < 4; ++p) {
      float a1 = ((p >> 1) == (q >> 1)) ? 0.5f * s0[p] : 0.0f;  // M*S1
      A2[q][p] = s1[q] * a1;                                    // S2*M*S1
    }
  for (int q = 0; q < 4; ++q)
    for (int p = 0; p < 4; ++p)
      T[q][p] = 0.5f * (A2[q & ~1][p] + A2[q | 1][p]);          // M*S2*M*S1

  int tid = blockIdx.x * blockDim.x + threadIdx.x;
  for (int idx = tid; idx < COUT * KTOT; idx += gridDim.x * blockDim.x) {
    int o = idx / KTOT, k = idx - o * KTOT;
    int gi = k >> 2, slot = k & 3;
    float val = 0.0f;
    if (slot < 3) {
      int c = gi / 3, kh = gi - 3 * c, kw = slot;
      for (int p = 0; p < 4; ++p) {
        float u = 0.0f;
        for (int q = 0; q < 4; ++q) u += w_post[o*4+q] * T[q][p];
        val += u * w_pre[((p * CIN + c) * 3 + kh) * 3 + kw];
      }
    }
    Wb[idx] = f2bf(val);
  }
}

// ---------------------------------------------------------------------------
// Kernel 2: fused implicit-GEMM 3x3 conv via bf16 WMMA.
// Block: 256 thr = 8 waves; each wave owns a unique 16-wide n-eighth (of 128) and
// loops over all 4 o-tiles, so each B fragment is gathered from LDS exactly
// once per chunk. B gather = one base pointer + literal offsets so the
// backend can fuse pairs into ds_load_2addr_b32.
// ---------------------------------------------------------------------------
__global__ __launch_bounds__(256) void conv_kernel(
    const float* __restrict__ x, const u16* __restrict__ Wb,
    const float* __restrict__ bias, float* __restrict__ out,
    float* __restrict__ partial) {
  extern __shared__ char smem[];
  u16*  xs  = (u16*)smem;
  const u32* xsw = (const u32*)smem;
  float* ps = (float*)(smem + PS_OFF);   // [sum: 64*8] + [sq: 64*8]

  const int tid = threadIdx.x;
  const int wt = blockIdx.x, h = blockIdx.y, b = blockIdx.z;
  const int w0 = wt * WT;

  // --- stage input tile to LDS as bf16, two parity copies (vectorized) ---
  const float* xb = x + (size_t)b * CIN * HWIN * HWIN;
  for (int ii = tid; ii < NGI * 33; ii += 256) {
    int gi = ii / 33, blk = ii - gi * 33;
    int c = gi / 3, kh = gi - 3 * c;
    int j = blk * 4;
    const float* rowp = xb + (size_t)c * (HWIN*HWIN) + (h + kh) * HWIN;
    int gw4 = w0 + j; if (gw4 > 252) gw4 = 252;      // in-bounds block clamp
    float4 v4 = *(const float4*)(rowp + gw4);
    int gwe = w0 + j + 4; if (gwe > 255) gwe = 255;  // halo element
    float e4 = rowp[gwe];
    u16 c0 = f2bf(v4.x), c1 = f2bf(v4.y), c2 = f2bf(v4.z), c3 = f2bf(v4.w);
    u16 c4 = f2bf(e4);
    u32* p0 = (u32*)(xs + gi * PITCH + j);
    p0[0] = (u32)c0 | ((u32)c1 << 16);
    p0[1] = (u32)c2 | ((u32)c3 << 16);
    u32* p1 = (u32*)(xs + XS1 + gi * PITCH + j);     // shifted copy
    p1[0] = (u32)c1 | ((u32)c2 << 16);
    p1[1] = (u32)c3 | ((u32)c4 << 16);
  }
  __syncthreads();

  const int wave = tid >> 5, lane = tid & 31;
  const int l16 = lane & 15, gh = lane >> 4;
  const int n = wave * 16 + l16;        // this wave's spatial column, 0..127

  v8f zero = {};
  v8f acc[4];
  #pragma unroll
  for (int t = 0; t < 4; ++t) acc[t] = zero;

  const int4* W4 = (const int4*)Wb;

  // lane-constant base for B gathers (parity-selected copy), even ushort idx
  const int par = n & 1;
  const int lanebase = (par ? XS1 : 0) + 4 * gh * PITCH + (n - par);
  const u32* bbase = xsw + (lanebase >> 1);

  for (int kc = 0; kc < CHUNKS; ++kc) {
    int kbase = kc * 32;
    const u32* bp = bbase + kc * (8 * PITCH / 2);   // advance 8 rows / chunk

    // B fragment: base + literal offsets -> ds_load_2addr candidates
    v8i bv;
    bv[0] = bp[0];           bv[1] = bp[1];
    bv[2] = bp[PITCH/2];     bv[3] = bp[PITCH/2 + 1];
    bv[4] = bp[PITCH];       bv[5] = bp[PITCH + 1];
    bv[6] = bp[3*PITCH/2];   bv[7] = bp[3*PITCH/2 + 1];
    v16bf Bf = __builtin_bit_cast(v16bf, bv);

    #pragma unroll
    for (int ot = 0; ot < 4; ++ot) {
      int oRow = ot * 16 + l16;             // A-matrix row this lane carries
      int4 alo = W4[(oRow * KTOT + kbase + 16 * gh) >> 3];
      int4 ahi = W4[(oRow * KTOT + kbase + 16 * gh + 8) >> 3];
      v8i av;
      av[0]=alo.x; av[1]=alo.y; av[2]=alo.z; av[3]=alo.w;
      av[4]=ahi.x; av[5]=ahi.y; av[6]=ahi.z; av[7]=ahi.w;
      v16bf A = __builtin_bit_cast(v16bf, av);
      acc[ot] = __builtin_amdgcn_wmma_f32_16x16x32_bf16(
          false, A, false, Bf, (short)0, acc[ot], false, false);
    }
  }

  // --- epilogue: bias, store pre-BN output, accumulate channel stats ---
  const int w = w0 + n;
  const bool ok = (w < WO);

  #pragma unroll
  for (int ot = 0; ot < 4; ++ot) {
    const float* bp2 = bias + ot * 16 + gh * 8;
    float4 b0 = ((const float4*)bp2)[0];
    float4 b1 = ((const float4*)bp2)[1];
    float bb[8] = {b0.x, b0.y, b0.z, b0.w, b1.x, b1.y, b1.z, b1.w};

    float y[8], ls[8], ls2[8];
    #pragma unroll
    for (int r = 0; r < 8; ++r) {
      y[r] = acc[ot][r] + bb[r];
      float ym = ok ? y[r] : 0.0f;          // branch-free masked accumulate
      ls[r] = ym; ls2[r] = ym * ym;
    }
    if (ok) {
      // one base address; 8 row stores at constant offsets r*PLANE
      float* op = out + (((size_t)b * COUT + ot * 16 + gh * 8) * HO + h) * WO + w;
      #pragma unroll
      for (int r = 0; r < 8; ++r) op[(size_t)r * PLANE] = y[r];
    }
    #pragma unroll
    for (int m = 1; m < 16; m <<= 1) {
      #pragma unroll
      for (int r = 0; r < 8; ++r) {
        ls[r]  += __shfl_xor(ls[r],  m, 32);
        ls2[r] += __shfl_xor(ls2[r], m, 32);
      }
    }
    if (l16 == 0) {
      #pragma unroll
      for (int r = 0; r < 8; ++r) {
        int o = ot * 16 + gh * 8 + r;
        ps[o * 8 + wave]       = ls[r];
        ps[512 + o * 8 + wave] = ls2[r];
      }
    }
  }
  __syncthreads();
  if (tid < 64) {
    float s = 0.0f, s2 = 0.0f;
    #pragma unroll
    for (int wv = 0; wv < 8; ++wv) {
      s  += ps[tid * 8 + wv];
      s2 += ps[512 + tid * 8 + wv];
    }
    size_t bid = ((size_t)b * HO + h) * 2 + wt;
    partial[bid * 128 + tid]      = s;
    partial[bid * 128 + 64 + tid] = s2;
  }
}

// ---------------------------------------------------------------------------
// Kernel 3: reduce per-block partials -> per-channel scale/shift.
// ---------------------------------------------------------------------------
__global__ void stats_kernel(const float* __restrict__ partial,
                             const float* __restrict__ gamma,
                             const float* __restrict__ beta,
                             float* __restrict__ coef) {
  __shared__ float sh[256], sh2[256];
  int o = blockIdx.x, t = threadIdx.x;
  float s = 0.0f, s2 = 0.0f;
  for (int i = t; i < NBLK; i += 256) {
    s  += partial[(size_t)i * 128 + o];
    s2 += partial[(size_t)i * 128 + 64 + o];
  }
  sh[t] = s; sh2[t] = s2; __syncthreads();
  for (int st = 128; st > 0; st >>= 1) {
    if (t < st) { sh[t] += sh[t + st]; sh2[t] += sh2[t + st]; }
    __syncthreads();
  }
  if (t == 0) {
    float invN = 1.0f / (float)NPIX;
    float mean = sh[0] * invN;
    float var  = sh2[0] * invN - mean * mean;
    float sc = gamma[o] * rsqrtf(var + 1e-5f);
    coef[o]      = sc;
    coef[64 + o] = beta[o] - mean * sc;
  }
}

// ---------------------------------------------------------------------------
// Kernel 4: in-place normalize + ReLU (float4, plane size % 4 == 0).
// ---------------------------------------------------------------------------
__global__ void norm_kernel(float* __restrict__ out,
                            const float* __restrict__ coef) {
  __shared__ float sc[64], sf[64];
  if (threadIdx.x < 64) {
    sc[threadIdx.x] = coef[threadIdx.x];
    sf[threadIdx.x] = coef[64 + threadIdx.x];
  }
  __syncthreads();
  float4* o4 = (float4*)out;
  size_t total4 = (size_t)BATCH * COUT * PLANE / 4;
  for (size_t i = (size_t)blockIdx.x * 256 + threadIdx.x; i < total4;
       i += (size_t)gridDim.x * 256) {
    float4 v = o4[i];
    int o = (int)((i * 4) / PLANE) & 63;
    float a = sc[o], c = sf[o];
    v.x = fmaxf(v.x * a + c, 0.0f);
    v.y = fmaxf(v.y * a + c, 0.0f);
    v.z = fmaxf(v.z * a + c, 0.0f);
    v.w = fmaxf(v.w * a + c, 0.0f);
    o4[i] = v;
  }
}

extern "C" void kernel_launch(void* const* d_in, const int* in_sizes, int n_in,
                              void* d_out, int out_size, void* d_ws, size_t ws_size,
                              hipStream_t stream) {
  const float* x      = (const float*)d_in[0];
  const float* w_pre  = (const float*)d_in[1];
  const float* qp     = (const float*)d_in[2];
  const float* w_post = (const float*)d_in[3];
  const float* b_post = (const float*)d_in[4];
  const float* gamma  = (const float*)d_in[5];
  const float* beta   = (const float*)d_in[6];
  float* out = (float*)d_out;

  char* ws = (char*)d_ws;
  u16*   Wb      = (u16*)ws;                         // 64*768*2   = 96 KB
  float* coef    = (float*)(ws + COUT * KTOT * 2);   // 128 floats
  float* partial = (float*)(ws + COUT * KTOT * 2 + 1024);  // 8128*128*4 ~ 4.2 MB

  prep_kernel<<<64, 256, 0, stream>>>(w_pre, qp, w_post, Wb);

  dim3 g(2, HO, BATCH);
  conv_kernel<<<g, 256, SMEM_BYTES, stream>>>(x, Wb, b_post, out, partial);

  stats_kernel<<<COUT, 256, 0, stream>>>(partial, gamma, beta, coef);

  norm_kernel<<<4096, 256, 0, stream>>>(out, coef);
}
